// TAGConv_5179730559346
// MI455X (gfx1250) — matrix operand order repaired and checked by reading, not compiled, not run on hardware
//
#include <hip/hip_runtime.h>
#include <stdint.h>

// ---------------- problem constants ----------------
#define NNODES 50000
#define NEDGES 800000
#define DIN    128
#define DOUT   128
#define KHOP   2
#define KDIM   (DIN * (KHOP + 1))   // 384

typedef __attribute__((ext_vector_type(16))) __bf16 v16bf;
typedef __attribute__((ext_vector_type(8)))  __bf16 v8bf;
typedef __attribute__((ext_vector_type(8)))  float  v8f;

__device__ __forceinline__ unsigned short f2bf(float f) {
    unsigned int u = __float_as_uint(f);
    unsigned int r = u + 0x7FFFu + ((u >> 16) & 1u);   // round-to-nearest-even
    return (unsigned short)(r >> 16);
}

// ---------------- degree count: one thread per edge ----------------
__global__ __launch_bounds__(256) void deg_kernel(const int* __restrict__ dst,
                                                  float* __restrict__ deg, int E) {
    int e = blockIdx.x * blockDim.x + threadIdx.x;
    if (e >= E) return;
    atomicAdd(&deg[dst[e]], 1.0f);
}

// ---------------- invdeg[i] = 1 / max(deg,1) (in place) ----------------
__global__ __launch_bounds__(256) void invdeg_kernel(float* __restrict__ deg, int n) {
    int i = blockIdx.x * blockDim.x + threadIdx.x;
    if (i >= n) return;
    float d = deg[i];
    deg[i] = 1.0f / fmaxf(d, 1.0f);
}

// ---------------- scatter-add: one wave per edge, 4 floats per lane ----------------
__global__ __launch_bounds__(256) void scatter_kernel(const float* __restrict__ xin,
                                                      const int* __restrict__ src,
                                                      const int* __restrict__ dst,
                                                      float* __restrict__ sums, int E) {
    int t = blockIdx.x * blockDim.x + threadIdx.x;
    int e = t >> 5;                  // wave32: one edge per wave
    if (e >= E) return;
    int c = (t & 31) << 2;           // 32 lanes x 4 floats = 128 features
    int s = src[e];
    int d = dst[e];
    const float4 v = *(const float4*)(xin + (size_t)s * DIN + c);
    float* p = sums + (size_t)d * DIN + c;
    atomicAdd(p + 0, v.x);
    atomicAdd(p + 1, v.y);
    atomicAdd(p + 2, v.z);
    atomicAdd(p + 3, v.w);
}

// ---- normalize sums by invdeg; write f32 copy (optional) + bf16 slice of h ----
__global__ __launch_bounds__(256) void normalize_kernel(const float* __restrict__ sums,
                                                        const float* __restrict__ invdeg,
                                                        float* __restrict__ pout,       // may be null
                                                        unsigned short* __restrict__ hbf,
                                                        int slice, int n) {
    int t = blockIdx.x * blockDim.x + threadIdx.x;       // n * DIN threads
    if (t >= n * DIN) return;
    int i = t >> 7;                  // node
    int f = t & (DIN - 1);           // feature
    float v = sums[(size_t)i * DIN + f] * invdeg[i];
    if (pout) pout[(size_t)i * DIN + f] = v;
    hbf[(size_t)i * KDIM + slice * DIN + f] = f2bf(v);
}

// ---------------- f32 -> bf16 copy of x into slice 0 of h ----------------
__global__ __launch_bounds__(256) void cvt_x_kernel(const float* __restrict__ x,
                                                    unsigned short* __restrict__ hbf, int n) {
    int t = blockIdx.x * blockDim.x + threadIdx.x;
    if (t >= n * DIN) return;
    int i = t >> 7;
    int f = t & (DIN - 1);
    hbf[(size_t)i * KDIM + f] = f2bf(x[t]);
}

// ---------------- f32 -> bf16 copy of W ----------------
__global__ __launch_bounds__(256) void cvt_w_kernel(const float* __restrict__ W,
                                                    unsigned short* __restrict__ Wbf, int n) {
    int t = blockIdx.x * blockDim.x + threadIdx.x;
    if (t >= n) return;
    Wbf[t] = f2bf(W[t]);
}

// ---------------- WMMA GEMM: out[N,128] = h_bf16[N,384] @ Wbf.T + b ----------------
// One block = 256 threads = 8 waves. Block handles a 16-row tile of nodes;
// each wave owns one 16-wide output-column tile (8 * 16 = 128 = DOUT).
__global__ __launch_bounds__(256) void gemm_wmma_kernel(const unsigned short* __restrict__ hbf,
                                                        const unsigned short* __restrict__ Wbf,
                                                        const float* __restrict__ bias,
                                                        float* __restrict__ out) {
    const int lane = threadIdx.x & 31;
    const int wave = threadIdx.x >> 5;          // 0..7 -> column tile
    const int row0 = blockIdx.x * 16;           // N = 50000 = 3125 * 16 exactly
    const int col0 = wave * 16;

    const int half = lane >> 4;                 // 0: lanes 0-15, 1: lanes 16-31
    const int l    = lane & 15;

    // A layout (16x32 bf16): lane l<16 -> row l, K = kb..kb+15
    //                        lane l>=16 -> row l, K = kb+16..kb+31
    const __bf16* arow = (const __bf16*)hbf + (size_t)(row0 + l) * KDIM + half * 16;
    // B layout (32x16 bf16): lane l<16 -> col l, K = kb..kb+15 ; lanes 16-31 -> K = kb+16..kb+31
    const __bf16* brow = (const __bf16*)Wbf + (size_t)(col0 + l) * KDIM + half * 16;

    v8f acc = {};
    #pragma unroll
    for (int kb = 0; kb < KDIM; kb += 32) {
        v8bf alo = *(const v8bf*)(arow + kb);
        v8bf ahi = *(const v8bf*)(arow + kb + 8);
        v16bf A, B;
        {
            v16bf bb = *(const v16bf*)(brow + kb);
            B = bb;
        }
        #pragma unroll
        for (int i = 0; i < 8; ++i) { A[i] = alo[i]; A[i + 8] = ahi[i]; }
        acc = __builtin_amdgcn_wmma_f32_16x16x32_bf16(
            /*neg_a=*/false, A, /*neg_b=*/false, B,
            /*c_mod=*/(short)0, acc, /*reuse_a=*/false, /*reuse_b=*/false);
    }

    // C/D layout: VGPR v -> row (row0 + half*8 + v), col (col0 + l)
    const float bc = bias[col0 + l];
    float* obase = out + (size_t)(row0 + half * 8) * DOUT + col0 + l;
    #pragma unroll
    for (int v = 0; v < 8; ++v) {
        obase[(size_t)v * DOUT] = acc[v] + bc;
    }
}

// ---------------- host-side orchestration ----------------
extern "C" void kernel_launch(void* const* d_in, const int* in_sizes, int n_in,
                              void* d_out, int out_size, void* d_ws, size_t ws_size,
                              hipStream_t stream) {
    const float* x  = (const float*)d_in[0];           // [N, 128]
    const int*   ei = (const int*)d_in[1];             // [2, E] flattened
    const float* W  = (const float*)d_in[2];           // [128, 384]
    const float* b  = (const float*)d_in[3];           // [128]
    float* out = (float*)d_out;                        // [N, 128]

    const int* src = ei;
    const int* dst = ei + NEDGES;

    // workspace carve-up (bytes, 256-aligned)
    char* ws = (char*)d_ws;
    size_t off = 0;
    auto carve = [&](size_t bytes) { char* p = ws + off; off = (off + bytes + 255) & ~(size_t)255; return p; };
    float*          deg  = (float*)carve((size_t)NNODES * sizeof(float));
    float*          sums = (float*)carve((size_t)NNODES * DIN * sizeof(float));
    float*          p1   = (float*)carve((size_t)NNODES * DIN * sizeof(float));
    unsigned short* hbf  = (unsigned short*)carve((size_t)NNODES * KDIM * sizeof(unsigned short));
    unsigned short* Wbf  = (unsigned short*)carve((size_t)DOUT * KDIM * sizeof(unsigned short));
    (void)ws_size;

    const int TB = 256;
    const int edgeWarpBlocks = (NEDGES * 32 + TB - 1) / TB;
    const int featBlocks     = (NNODES * DIN + TB - 1) / TB;

    // degree -> invdeg
    hipMemsetAsync(deg, 0, (size_t)NNODES * sizeof(float), stream);
    deg_kernel<<<(NEDGES + TB - 1) / TB, TB, 0, stream>>>(dst, deg, NEDGES);
    invdeg_kernel<<<(NNODES + TB - 1) / TB, TB, 0, stream>>>(deg, NNODES);

    // slice 0 of h (= x) and bf16 W, independent of propagation
    cvt_x_kernel<<<featBlocks, TB, 0, stream>>>(x, hbf, NNODES);
    cvt_w_kernel<<<(DOUT * KDIM + TB - 1) / TB, TB, 0, stream>>>(W, Wbf, DOUT * KDIM);

    // hop 1: sums = scatter_add(x), p1 = sums * invdeg, slice 1 of h
    hipMemsetAsync(sums, 0, (size_t)NNODES * DIN * sizeof(float), stream);
    scatter_kernel<<<edgeWarpBlocks, TB, 0, stream>>>(x, src, dst, sums, NEDGES);
    normalize_kernel<<<featBlocks, TB, 0, stream>>>(sums, deg, p1, hbf, 1, NNODES);

    // hop 2: sums = scatter_add(p1), slice 2 of h (no f32 copy needed)
    hipMemsetAsync(sums, 0, (size_t)NNODES * DIN * sizeof(float), stream);
    scatter_kernel<<<edgeWarpBlocks, TB, 0, stream>>>(p1, src, dst, sums, NEDGES);
    normalize_kernel<<<featBlocks, TB, 0, stream>>>(sums, deg, (float*)nullptr, hbf, 2, NNODES);

    // fused linear: WMMA bf16 GEMM + bias
    gemm_wmma_kernel<<<NNODES / 16, TB, 0, stream>>>(hbf, Wbf, b, out);
}
